// AttentionBlock_82806969467101
// MI455X (gfx1250) — matrix operand order repaired and checked by reading, not compiled, not run on hardware
//
#include <hip/hip_runtime.h>
#include <math.h>

typedef _Float16 half_t;
typedef __attribute__((ext_vector_type(16))) _Float16 v16h;
typedef __attribute__((ext_vector_type(8)))  _Float16 v8h;
typedef __attribute__((ext_vector_type(4)))  _Float16 v4h;
typedef __attribute__((ext_vector_type(8)))  float    v8f;
typedef __attribute__((ext_vector_type(4)))  float    v4f;
typedef unsigned int u32;
typedef __attribute__((ext_vector_type(4)))  u32      v4u;
typedef __attribute__((ext_vector_type(8)))  u32      v8u;

constexpr int BATCH = 32;
constexpr int C     = 512;
constexpr int HW    = 1024;   // tokens per image
constexpr int O3    = 1536;   // 3*C
constexpr int CPG   = 16;     // channels per group (512/32)
constexpr float EPSV = 1e-5f;
// 1/sqrt(sqrt(512)) baked into both q and k
#define ATT_SCALE 0.21022410381342864f

// ---------------------------------------------------------------------------
// A-fragment load for 16-bit A 16x32 layout:
// lanes 0-15 (row m) hold K 0-7 & 16-23; lanes 16-31 (row m) hold K 8-15 & 24-31
// -> two contiguous v8h (b128) loads per lane.
// ---------------------------------------------------------------------------
__device__ __forceinline__ v16h load_a16(const half_t* __restrict__ p) {
    v8h lo = *(const v8h*)p;
    v8h hi = *(const v8h*)(p + 16);
    v16h a;
#pragma unroll
    for (int t = 0; t < 8; ++t) { a[t] = lo[t]; a[t + 8] = hi[t]; }
    return a;
}

// ---------------------------------------------------------------------------
// 2x2 register-blocked WMMA macro-tile: one wave computes a 32x32 output block.
// A fragments reused across 2 N-subtiles, B fragments across 2 M-subtiles
// -> 2 b128 loads per v_wmma instead of 4.
// A: row-major [M][K] (lda), B: row-major [N][K] (ldb), f32 accumulate.
// ---------------------------------------------------------------------------
template <int K>
__device__ __forceinline__ void wmma_macro_2x2(const half_t* __restrict__ Abase, int lda,
                                               const half_t* __restrict__ Bbase, int ldb,
                                               v8f acc[2][2]) {
    const int lane = threadIdx.x & 31;
    const int hs   = lane >> 4;       // half-wave select
    const int r    = lane & 15;
    const half_t* a0 = Abase + (size_t)r * lda + hs * 8;
    const half_t* a1 = a0 + (size_t)16 * lda;
    const half_t* b0 = Bbase + (size_t)r * ldb + hs * 8;
    const half_t* b1 = b0 + (size_t)16 * ldb;

    acc[0][0] = (v8f){}; acc[0][1] = (v8f){};
    acc[1][0] = (v8f){}; acc[1][1] = (v8f){};

#pragma unroll 2
    for (int k0 = 0; k0 < K; k0 += 32) {
        if ((k0 & 63) == 0) {   // statically resolved under unroll-2
            __builtin_prefetch(a0 + k0 + 256, 0, 1);   // global_prefetch_b8
            __builtin_prefetch(a1 + k0 + 256, 0, 1);
            __builtin_prefetch(b0 + k0 + 256, 0, 1);
            __builtin_prefetch(b1 + k0 + 256, 0, 1);
        }
        v16h A0 = load_a16(a0 + k0);
        v16h A1 = load_a16(a1 + k0);
        v16h B0 = load_a16(b0 + k0);
        v16h B1 = load_a16(b1 + k0);
        acc[0][0] = __builtin_amdgcn_wmma_f32_16x16x32_f16(false, A0, false, B0,
                                                           (short)0, acc[0][0], false, false);
        acc[0][1] = __builtin_amdgcn_wmma_f32_16x16x32_f16(false, A0, false, B1,
                                                           (short)0, acc[0][1], false, false);
        acc[1][0] = __builtin_amdgcn_wmma_f32_16x16x32_f16(false, A1, false, B0,
                                                           (short)0, acc[1][0], false, false);
        acc[1][1] = __builtin_amdgcn_wmma_f32_16x16x32_f16(false, A1, false, B1,
                                                           (short)0, acc[1][1], false, false);
    }
}

// ---------------------------------------------------------------------------
// Weight conversion f32 -> f16 (qkv_w 1536x512 then proj_w 512x512)
// ---------------------------------------------------------------------------
__global__ __launch_bounds__(256) void convert_weights_kernel(
    const float* __restrict__ qw, const float* __restrict__ pw,
    half_t* __restrict__ Wq, half_t* __restrict__ Wp) {
    int i = blockIdx.x * 256 + threadIdx.x;            // 0 .. 1048575
    if (i < O3 * C) Wq[i] = (half_t)qw[i];
    else            Wp[i - O3 * C] = (half_t)pw[i - O3 * C];
}

// ---------------------------------------------------------------------------
// GroupNorm: one block per (batch, group). 16 channels x 1024 pixels.
// The x slab (16x1024 f32 = 64 KB) is DMA'd into LDS ONCE by the Tensor Data
// Mover (tensor_load_to_lds, TENSORcnt), so both the statistics pass and the
// normalization pass read LDS instead of touching HBM twice.
// Normalized tile staged in LDS, then written transposed h_t[b][n][c]
// as 32-byte contiguous stores per lane.
// ---------------------------------------------------------------------------
__global__ __launch_bounds__(256) void groupnorm_kernel(
    const float* __restrict__ x, const float* __restrict__ gamma,
    const float* __restrict__ beta, half_t* __restrict__ h_t) {
    const int bg = blockIdx.x;          // 0..1023
    const int b  = bg >> 5;
    const int g  = bg & 31;
    const int t  = threadIdx.x;
    const float* xg = x + ((size_t)b * C + g * CPG) * HW;   // 16 x 1024 slab

    constexpr int LPITCH = HW + 8;      // pad to break bank alignment
    __shared__ float  xs[CPG * HW];     // 64 KB DMA target
    __shared__ half_t tile[CPG * LPITCH];
    __shared__ float  red[256];

    // ---- TDM: issue one tensor DMA per workgroup (wave 0 only; TDM ignores
    // EXEC and issues once per wave). D# per CDNA5 ISA ch.8.
    if ((t >> 5) == 0) {
        const unsigned long long gaddr = (unsigned long long)(uintptr_t)xg;
        const u32 lds_addr = (u32)(uintptr_t)&xs[0];   // low 32 bits = LDS offset
        v4u g0;
        g0[0] = 1u;                                    // count=1 (valid), user mode
        g0[1] = lds_addr;                              // lds_addr
        g0[2] = (u32)gaddr;                            // global_addr[31:0]
        g0[3] = (u32)(gaddr >> 32) | (2u << 30);       // global_addr[56:32] | type=2
        v8u g1;
        g1[0] = 2u << 16;                              // data_size=4B; no multicast/pad
        g1[1] = (u32)HW << 16;                         // tensor_dim0 = 1024 (lo16)
        g1[2] = (u32)CPG << 16;                        // tensor_dim0 hi=0 | tensor_dim1=16
        g1[3] = (u32)HW << 16;                         // tensor_dim1 hi=0 | tile_dim0=1024
        g1[4] = (u32)CPG;                              // tile_dim1=16 | tile_dim2=0
        g1[5] = (u32)HW;                               // tensor_dim0_stride = 1024 (lo32)
        g1[6] = 0u;                                    // stride hi | dim1_stride lo
        g1[7] = 0u;                                    // dim1_stride hi
        asm volatile("tensor_load_to_lds %0, %1"
                     :: "s"(g0), "s"(g1)
                     : "memory");
        __builtin_amdgcn_s_wait_tensorcnt(0);
    }
    __syncthreads();                                   // LDS tile visible to all waves

    // ---- statistics from LDS (b128 ds loads)
    float s = 0.f, ss = 0.f;
#pragma unroll
    for (int i = 0; i < 16; ++i) {
        v4f v = *(const v4f*)(xs + 4 * t + HW * i);
        s  += v[0] + v[1] + v[2] + v[3];
        ss += v[0] * v[0] + v[1] * v[1] + v[2] * v[2] + v[3] * v[3];
    }
    red[t] = s; __syncthreads();
    for (int off = 128; off > 0; off >>= 1) {
        if (t < off) red[t] += red[t + off];
        __syncthreads();
    }
    const float mean = red[0] * (1.0f / 16384.0f);
    __syncthreads();
    red[t] = ss; __syncthreads();
    for (int off = 128; off > 0; off >>= 1) {
        if (t < off) red[t] += red[t + off];
        __syncthreads();
    }
    const float var  = red[0] * (1.0f / 16384.0f) - mean * mean;
    const float rstd = rsqrtf(var + EPSV);

    // ---- normalize from LDS into padded f16 tile (row uniform per iteration)
#pragma unroll
    for (int i = 0; i < 16; ++i) {
        const int c = g * CPG + i;
        const float ga = gamma[c] * rstd;
        const float be = beta[c] - mean * ga;
        v4f v = *(const v4f*)(xs + 4 * t + HW * i);
        v4h o;
        o[0] = (half_t)(v[0] * ga + be); o[1] = (half_t)(v[1] * ga + be);
        o[2] = (half_t)(v[2] * ga + be); o[3] = (half_t)(v[3] * ga + be);
        *(v4h*)(tile + i * LPITCH + 4 * t) = o;
    }
    __syncthreads();

    // ---- transpose out of LDS: each thread emits 16 f16 = one 32B store
#pragma unroll
    for (int i = 0; i < 4; ++i) {
        int n = t + 256 * i;
        v16h v;
#pragma unroll
        for (int cl = 0; cl < CPG; ++cl) v[cl] = tile[cl * LPITCH + n];
        *(v16h*)(h_t + ((size_t)b * HW + n) * C + g * CPG) = v;
    }
}

// ---------------------------------------------------------------------------
// QKV GEMM: M=1024 (tokens), N=1536 (outputs), K=512.  32x32 tile per wave.
// A = h_t[b] (i x c), B = Wq rows (o x c).  q,k stored [i][o'] pre-scaled;
// v stored [o'][i] via transposed (coalesced) store.
// ---------------------------------------------------------------------------
__global__ __launch_bounds__(256) void gemm_qkv_kernel(
    const half_t* __restrict__ h_t, const half_t* __restrict__ Wq,
    const float* __restrict__ qkv_b,
    half_t* __restrict__ q_t, half_t* __restrict__ k_t, half_t* __restrict__ v_m) {
    const int wave = blockIdx.x * 8 + (threadIdx.x >> 5);
    const int nt = wave % (O3 / 32);                 // 48
    const int mt = (wave / (O3 / 32)) % (HW / 32);   // 32
    const int b  = wave / ((O3 / 32) * (HW / 32));
    const int m0 = mt * 32, n0 = nt * 32;

    const half_t* A = h_t + (size_t)b * HW * C + (size_t)m0 * C;
    const half_t* B = Wq + (size_t)n0 * C;
    v8f acc[2][2];
    wmma_macro_2x2<C>(A, C, B, C, acc);

    const int lane = threadIdx.x & 31;
    const int hs = lane >> 4, nl = lane & 15;
#pragma unroll
    for (int tj = 0; tj < 2; ++tj) {
        const int o = n0 + 16 * tj + nl;
        const float bias = qkv_b[o];
        if (o < 2 * C) {   // q or k -> [i][o'] layout, pre-scaled
            half_t* dst = (o < C) ? (q_t + (size_t)b * HW * C + o)
                                  : (k_t + (size_t)b * HW * C + (o - C));
#pragma unroll
            for (int ti = 0; ti < 2; ++ti) {
                const int mrow = m0 + 16 * ti + 8 * hs;
#pragma unroll
                for (int r = 0; r < 8; ++r)
                    dst[(size_t)(mrow + r) * C] =
                        (half_t)((acc[ti][tj][r] + bias) * ATT_SCALE);
            }
        } else {           // v -> [c][j] layout, contiguous per-lane store
            half_t* dstv = v_m + (size_t)b * C * HW + (size_t)(o - 2 * C) * HW;
#pragma unroll
            for (int ti = 0; ti < 2; ++ti) {
                const int mrow = m0 + 16 * ti + 8 * hs;
#pragma unroll
                for (int r = 0; r < 8; ++r)
                    dstv[mrow + r] = (half_t)(acc[ti][tj][r] + bias);
            }
        }
    }
}

// ---------------------------------------------------------------------------
// Scores GEMM: w[i][j] = sum_c q_t[i][c] * k_t[j][c], f32 output.
// ---------------------------------------------------------------------------
__global__ __launch_bounds__(256) void gemm_scores_kernel(
    const half_t* __restrict__ q_t, const half_t* __restrict__ k_t,
    float* __restrict__ scores) {
    const int wave = blockIdx.x * 8 + (threadIdx.x >> 5);
    const int nt = wave % (HW / 32);
    const int mt = (wave / (HW / 32)) % (HW / 32);
    const int b  = wave / ((HW / 32) * (HW / 32));
    const int m0 = mt * 32, n0 = nt * 32;

    const half_t* A = q_t + (size_t)b * HW * C + (size_t)m0 * C;
    const half_t* B = k_t + (size_t)b * HW * C + (size_t)n0 * C;
    v8f acc[2][2];
    wmma_macro_2x2<C>(A, C, B, C, acc);

    const int lane = threadIdx.x & 31;
    const int hs = lane >> 4, nl = lane & 15;
    float* dst = scores + (size_t)b * HW * HW;
#pragma unroll
    for (int ti = 0; ti < 2; ++ti)
#pragma unroll
        for (int tj = 0; tj < 2; ++tj)
#pragma unroll
            for (int r = 0; r < 8; ++r)
                dst[(size_t)(m0 + 16 * ti + 8 * hs + r) * HW + n0 + 16 * tj + nl] =
                    acc[ti][tj][r];
}

// ---------------------------------------------------------------------------
// Softmax over each length-1024 row; converts the row to f16 in place
// (f16 row occupies the first half of the f32 row storage).
// ---------------------------------------------------------------------------
__global__ __launch_bounds__(256) void softmax_kernel(float* __restrict__ scores) {
    const int row = blockIdx.x;                    // b*1024 + i
    float* rp = scores + (size_t)row * HW;
    const int t = threadIdx.x;
    v4f v = *(const v4f*)(rp + 4 * t);             // one b128 load per lane

    __shared__ float red[256];
    red[t] = fmaxf(fmaxf(v[0], v[1]), fmaxf(v[2], v[3]));
    __syncthreads();
    for (int off = 128; off > 0; off >>= 1) {
        if (t < off) red[t] = fmaxf(red[t], red[t + off]);
        __syncthreads();
    }
    const float mx = red[0];
    __syncthreads();
    float e0 = __expf(v[0] - mx), e1 = __expf(v[1] - mx);
    float e2 = __expf(v[2] - mx), e3 = __expf(v[3] - mx);
    red[t] = e0 + e1 + e2 + e3;
    __syncthreads();
    for (int off = 128; off > 0; off >>= 1) {
        if (t < off) red[t] += red[t + off];
        __syncthreads();
    }
    const float inv = 1.0f / red[0];
    __syncthreads();   // all row reads complete before in-place f16 overwrite
    v4h o;
    o[0] = (half_t)(e0 * inv); o[1] = (half_t)(e1 * inv);
    o[2] = (half_t)(e2 * inv); o[3] = (half_t)(e3 * inv);
    *(v4h*)((half_t*)rp + 4 * t) = o;              // one b64 store per lane
}

// ---------------------------------------------------------------------------
// Apply GEMM: att_t[i][c] = sum_j s[i][j] * v[c][j].  K = 1024.
// A = f16 softmax rows (element stride 2048 = f32 row pitch), B = v_m rows.
// ---------------------------------------------------------------------------
__global__ __launch_bounds__(256) void gemm_apply_kernel(
    const half_t* __restrict__ s_f16, const half_t* __restrict__ v_m,
    half_t* __restrict__ att_t) {
    const int wave = blockIdx.x * 8 + (threadIdx.x >> 5);
    const int nt = wave % (C / 32);
    const int mt = (wave / (C / 32)) % (HW / 32);
    const int b  = wave / ((C / 32) * (HW / 32));
    const int m0 = mt * 32, n0 = nt * 32;

    const half_t* A = s_f16 + ((size_t)b * HW + m0) * (2 * HW);   // lda = 2048
    const half_t* B = v_m + (size_t)b * C * HW + (size_t)n0 * HW; // ldb = 1024
    v8f acc[2][2];
    wmma_macro_2x2<HW>(A, 2 * HW, B, HW, acc);

    const int lane = threadIdx.x & 31;
    const int hs = lane >> 4, nl = lane & 15;
    half_t* dst = att_t + (size_t)b * HW * C;
#pragma unroll
    for (int ti = 0; ti < 2; ++ti)
#pragma unroll
        for (int tj = 0; tj < 2; ++tj)
#pragma unroll
            for (int r = 0; r < 8; ++r)
                dst[(size_t)(m0 + 16 * ti + 8 * hs + r) * C + n0 + 16 * tj + nl] =
                    (half_t)acc[ti][tj][r];
}

// ---------------------------------------------------------------------------
// Proj GEMM + residual: out[b][o][i] = x + sum_c att_t[i][c]*Wp[o][c] + pb[o].
// Transposed store is fully coalesced (8 contiguous f32 per lane).
// ---------------------------------------------------------------------------
__global__ __launch_bounds__(256) void gemm_proj_kernel(
    const half_t* __restrict__ att_t, const half_t* __restrict__ Wp,
    const float* __restrict__ proj_b, const float* __restrict__ x,
    float* __restrict__ out) {
    const int wave = blockIdx.x * 8 + (threadIdx.x >> 5);
    const int nt = wave % (C / 32);
    const int mt = (wave / (C / 32)) % (HW / 32);
    const int b  = wave / ((C / 32) * (HW / 32));
    const int m0 = mt * 32, n0 = nt * 32;

    const half_t* A = att_t + (size_t)b * HW * C + (size_t)m0 * C;
    const half_t* B = Wp + (size_t)n0 * C;
    v8f acc[2][2];
    wmma_macro_2x2<C>(A, C, B, C, acc);

    const int lane = threadIdx.x & 31;
    const int hs = lane >> 4, nl = lane & 15;
#pragma unroll
    for (int tj = 0; tj < 2; ++tj) {
        const int o = n0 + 16 * tj + nl;
        const float pb = proj_b[o];
#pragma unroll
        for (int ti = 0; ti < 2; ++ti) {
            const size_t base = ((size_t)b * C + o) * HW + m0 + 16 * ti + 8 * hs;
#pragma unroll
            for (int r = 0; r < 8; ++r)
                out[base + r] = x[base + r] + acc[ti][tj][r] + pb;
        }
    }
}

// ---------------------------------------------------------------------------
extern "C" void kernel_launch(void* const* d_in, const int* in_sizes, int n_in,
                              void* d_out, int out_size, void* d_ws, size_t ws_size,
                              hipStream_t stream) {
    const float* x      = (const float*)d_in[0];
    const float* gamma  = (const float*)d_in[1];
    const float* beta   = (const float*)d_in[2];
    const float* qkv_w  = (const float*)d_in[3];
    const float* qkv_b  = (const float*)d_in[4];
    const float* proj_w = (const float*)d_in[5];
    const float* proj_b = (const float*)d_in[6];
    float* out = (float*)d_out;

    char* ws = (char*)d_ws;
    half_t* Wq  = (half_t*)ws;                                  // 1.5 MB
    half_t* Wp  = (half_t*)(ws + (size_t)O3 * C * 2);           // 0.5 MB
    half_t* h_t = (half_t*)(ws + (2ull  << 20));                // 32 MB (reused as att_t)
    half_t* q_t = (half_t*)(ws + (34ull << 20));                // 32 MB
    half_t* k_t = (half_t*)(ws + (66ull << 20));                // 32 MB
    half_t* v_m = (half_t*)(ws + (98ull << 20));                // 32 MB
    float*  sc  = (float*) (ws + (130ull << 20));               // 128 MB

    convert_weights_kernel<<<4096, 256, 0, stream>>>(qkv_w, proj_w, Wq, Wp);
    groupnorm_kernel<<<BATCH * 32, 256, 0, stream>>>(x, gamma, beta, h_t);
    // 32 b * 32 mt * 48 nt waves / 8 = 6144 blocks
    gemm_qkv_kernel<<<6144, 256, 0, stream>>>(h_t, Wq, qkv_b, q_t, k_t, v_m);
    // 32 * 32 * 32 / 8 = 4096 blocks
    gemm_scores_kernel<<<4096, 256, 0, stream>>>(q_t, k_t, sc);
    softmax_kernel<<<BATCH * HW, 256, 0, stream>>>(sc);
    // 32 * 32 * 16 / 8 = 2048 blocks
    gemm_apply_kernel<<<2048, 256, 0, stream>>>((const half_t*)sc, v_m, h_t);
    gemm_proj_kernel<<<2048, 256, 0, stream>>>(h_t, Wp, proj_b, x, out);
}
